// LSTMandStack_Reverser_14070312861729
// MI455X (gfx1250) — matrix operand order repaired and checked by reading, not compiled, not run on hardware
//
#include <hip/hip_runtime.h>
#include <hip/hip_bf16.h>

typedef __attribute__((ext_vector_type(16))) _Float16 v16h;
typedef __attribute__((ext_vector_type(8)))  float    v8f;

#define Hdim   256
#define Mdim   32
#define IND    131
#define OUTD   131
#define NB     32
#define TSTEPS 256
#define TENC   128
#define KTX    5            // padded input K tiles (160 >= 131)
#define KTR    9            // recurrent K tiles: r(32)+h(256) = 288
#define NG     1024
#define NTG    64
#define NTO    16           // Wo: N = 256
#define NTL    9            // Wlin: N = 144 (pad of 131)
#define NLPAD  144
#define CAP    257

// ---- workspace layout (bytes, all 32B aligned) ----
#define OFF_XPROJ 0ull
#define SZ_XPROJ  ((size_t)TSTEPS * NB * NG * 4)          // 32 MB
#define OFF_WXE   (OFF_XPROJ + SZ_XPROJ)
#define SZ_WX     ((size_t)KTX * NTG * 512 * 2)
#define OFF_WXD   (OFF_WXE + SZ_WX)
#define OFF_WRE   (OFF_WXD + SZ_WX)
#define SZ_WR     ((size_t)KTR * NTG * 512 * 2)
#define OFF_WRD   (OFF_WRE + SZ_WR)
#define OFF_WO    (OFF_WRD + SZ_WR)
#define SZ_WO     ((size_t)8 * NTO * 512 * 2)
#define OFF_WL    (OFF_WO + SZ_WO)
#define SZ_WL     ((size_t)8 * NTL * 512 * 2)
#define OFF_VBUF  (OFF_WL + SZ_WL)
#define SZ_VBUF   ((size_t)CAP * NB * Mdim * 4)
#define OFF_SBUF  (OFF_VBUF + SZ_VBUF)

__device__ __forceinline__ int a_kmap(int j, int lane) {
  // 16-bit A-matrix 16x32 layout (CDNA5 ISA §7.12.2)
  return (j & 7) + ((lane >> 4) << 3) + ((j >> 3) << 4);
}
__device__ __forceinline__ float sigmoidf_(float x) { return 1.0f / (1.0f + __expf(-x)); }

// ======================= weight pre-pack (f32 -> f16 B-fragment layout) =======================
__global__ void stkrnn_pack_weights(const float* __restrict__ Wih_e, const float* __restrict__ Whh_e,
                                    const float* __restrict__ Wih_d, const float* __restrict__ Whh_d,
                                    const float* __restrict__ Wo,    const float* __restrict__ Wlin,
                                    unsigned char* __restrict__ ws) {
  const int SZX = KTX * NTG * 512;
  const int SZR = KTR * NTG * 512;
  const int SZO = 8 * NTO * 512;
  const int SZL = 8 * NTL * 512;
  const int total = 2 * SZX + 2 * SZR + SZO + SZL;
  for (int idx = blockIdx.x * blockDim.x + threadIdx.x; idx < total; idx += gridDim.x * blockDim.x) {
    int local = idx;
    _Float16* dst; int ntt, kind;
    const float* A0 = nullptr; const float* A1 = nullptr;
    if (local < SZX)             { dst = (_Float16*)(ws + OFF_WXE); ntt = NTG; kind = 0; A0 = Wih_e; }
    else if ((local -= SZX) < SZX) { dst = (_Float16*)(ws + OFF_WXD); ntt = NTG; kind = 0; A0 = Wih_d; }
    else if ((local -= SZX) < SZR) { dst = (_Float16*)(ws + OFF_WRE); ntt = NTG; kind = 1; A0 = Wih_e; A1 = Whh_e; }
    else if ((local -= SZR) < SZR) { dst = (_Float16*)(ws + OFF_WRD); ntt = NTG; kind = 1; A0 = Wih_d; A1 = Whh_d; }
    else if ((local -= SZR) < SZO) { dst = (_Float16*)(ws + OFF_WO);  ntt = NTO; kind = 2; A0 = Wo; }
    else       { local -= SZO;       dst = (_Float16*)(ws + OFF_WL);  ntt = NTL; kind = 3; A0 = Wlin; }
    int j = local & 15, ln = (local >> 4) & 31, rest = local >> 9;
    int nt = rest % ntt, kt = rest / ntt;
    int k = kt * 32 + (j & 7) + ((ln >> 4) << 3) + ((j >> 3) << 4);   // B-fragment: k = (j&7) + 8*(lane>=16) + 16*(j>=8) within 32-k tile
    int n = nt * 16 + (ln & 15);
    float v = 0.0f;
    if (kind == 0)      { if (k < IND) v = A0[n * (IND + Mdim) + k]; }
    else if (kind == 1) { v = (k < Mdim) ? A0[n * (IND + Mdim) + IND + k] : A1[n * Hdim + (k - Mdim)]; }
    else if (kind == 2) { v = A0[n * Hdim + k]; }
    else                { if (n < OUTD) v = A0[n * Hdim + k]; }
    dst[local] = (_Float16)v;
  }
}

// ======================= Xproj: x_t @ Wih[:, :131]^T + bih + bhh  (all steps, parallel WMMA) ===
__global__ void stkrnn_xproj(const float* __restrict__ X, const float* __restrict__ Y,
                             const float* __restrict__ bih_e, const float* __restrict__ bhh_e,
                             const float* __restrict__ bih_d, const float* __restrict__ bhh_d,
                             unsigned char* __restrict__ ws) {
  int wid  = (blockIdx.x * blockDim.x + threadIdx.x) >> 5;
  int lane = threadIdx.x & 31;
  const int total_tiles = (TSTEPS * NB / 16) * NTG;       // 512 * 64
  if (wid >= total_tiles) return;
  int mtile = wid / NTG, nt = wid % NTG;
  int t = mtile >> 1;
  bool dec = (t >= TENC);
  const v16h* wx = (const v16h*)(ws + (dec ? OFF_WXD : OFF_WXE));
  const float* bi = dec ? bih_d : bih_e;
  const float* bh = dec ? bhh_d : bhh_e;
  int n = nt * 16 + (lane & 15);
  float bias = bi[n] + bh[n];
  v8f c; for (int r = 0; r < 8; ++r) c[r] = bias;
  // teacher-forced input row: enc t<128 -> X[t]; t==128 -> X[128]; t>=129 -> Y[t-129]
  int b_row = ((mtile & 1) << 4) + (lane & 15);
  const float* xrow = (t <= TENC) ? (X + ((size_t)t * NB + b_row) * IND)
                                  : (Y + ((size_t)(t - TENC - 1) * NB + b_row) * IND);
  for (int kt = 0; kt < KTX; ++kt) {
    v16h a;
    #pragma unroll
    for (int j = 0; j < 16; ++j) {
      int k = kt * 32 + a_kmap(j, lane);
      a[j] = (k < IND) ? (_Float16)xrow[k] : (_Float16)0.0f;
    }
    v16h bf = wx[(kt * NTG + nt) * 32 + lane];
    c = __builtin_amdgcn_wmma_f32_16x16x32_f16(false, a, false, bf, (short)0, c, false, false);
  }
  float* xp = (float*)(ws + OFF_XPROJ);
  int rowbase = mtile * 16 + ((lane >> 4) << 3);
  #pragma unroll
  for (int r = 0; r < 8; ++r)
    xp[(size_t)(rowbase + r) * NG + nt * 16 + (lane & 15)] = c[r];
}

// ======================= persistent sequential kernel: 256 recurrent steps ====================
__global__ void __launch_bounds__(1024) stkrnn_seq(
    const float* __restrict__ Wd, const float* __restrict__ bd,
    const float* __restrict__ Wu, const float* __restrict__ bu,
    const float* __restrict__ Wv, const float* __restrict__ bv,
    const float* __restrict__ bo, const float* __restrict__ blin,
    const float* __restrict__ h0, const float* __restrict__ c0,
    unsigned char* __restrict__ ws, float* __restrict__ out) {
  __shared__ __align__(32) unsigned char sh_pool[65536];  // gates f16 [32][1024] / oc f16 + logits f32
  __shared__ __align__(32) _Float16 sh_h[NB * Hdim];
  __shared__ float                  sh_c[NB * Hdim];
  __shared__ __align__(32) _Float16 sh_r[NB * Mdim];
  __shared__ __align__(32) _Float16 sh_frag[18 * 512];    // WMMA A fragments
  __shared__ float                  sh_du[2 * NB];

  const int tid = threadIdx.x, w = tid >> 5, lane = tid & 31;
  float* sbuf = (float*)(ws + OFF_SBUF);
  float* Vbuf = (float*)(ws + OFF_VBUF);
  const float* xproj = (const float*)(ws + OFF_XPROJ);

  // ---- init state (deterministic: re-done every launch) ----
  for (int i = tid; i < NB * Hdim; i += 1024) { sh_h[i] = (_Float16)h0[i & 255]; sh_c[i] = c0[i & 255]; }
  for (int i = tid; i < NB * Mdim; i += 1024) { sh_r[i] = (_Float16)0.0f; Vbuf[i] = 0.0f; }
  for (int i = tid; i < CAP * NB;  i += 1024) sbuf[i] = 0.0f;
  __syncthreads();

  for (int t = 0; t < TSTEPS; ++t) {
    const bool dec = (t >= TENC);
    // ---- phase 1: pack A = [r | h] into WMMA A-fragments (2 mtiles x 9 ktiles) ----
    for (int idx = tid; idx < 18 * 512; idx += 1024) {
      int fr = idx >> 9, e = idx & 511, ln = e >> 4, j = e & 15;
      int mt = fr / KTR, kt = fr % KTR;
      int m = mt * 16 + (ln & 15);
      int k = kt * 32 + a_kmap(j, ln);
      sh_frag[idx] = (k < Mdim) ? sh_r[m * Mdim + k] : sh_h[m * Hdim + (k - Mdim)];
    }
    __syncthreads();
    // ---- phase 2: gate GEMM  g = Xproj[t] + [r|h] @ [Wih_r;Whh]  (wave w -> ntiles 2w,2w+1) ----
    {
      const v16h* wr = (const v16h*)(ws + (dec ? OFF_WRD : OFF_WRE));
      const v16h* af = (const v16h*)sh_frag;
      const int nt0 = w * 2, nt1 = w * 2 + 1;
      const int rb = (lane >> 4) << 3;
      v8f c00, c01, c10, c11;
      #pragma unroll
      for (int r = 0; r < 8; ++r) {
        size_t row0 = (size_t)t * NB + r + rb, row1 = row0 + 16;
        c00[r] = xproj[row0 * NG + nt0 * 16 + (lane & 15)];
        c01[r] = xproj[row0 * NG + nt1 * 16 + (lane & 15)];
        c10[r] = xproj[row1 * NG + nt0 * 16 + (lane & 15)];
        c11[r] = xproj[row1 * NG + nt1 * 16 + (lane & 15)];
      }
      for (int kt = 0; kt < KTR; ++kt) {
        if (kt + 1 < KTR) __builtin_prefetch(&wr[((kt + 1) * NTG + nt0) * 32 + lane], 0, 1);
        v16h a0 = af[(0 * KTR + kt) * 32 + lane];
        v16h a1 = af[(1 * KTR + kt) * 32 + lane];
        v16h b0 = wr[(kt * NTG + nt0) * 32 + lane];
        v16h b1 = wr[(kt * NTG + nt1) * 32 + lane];
        c00 = __builtin_amdgcn_wmma_f32_16x16x32_f16(false, a0, false, b0, (short)0, c00, false, false);
        c01 = __builtin_amdgcn_wmma_f32_16x16x32_f16(false, a0, false, b1, (short)0, c01, false, false);
        c10 = __builtin_amdgcn_wmma_f32_16x16x32_f16(false, a1, false, b0, (short)0, c10, false, false);
        c11 = __builtin_amdgcn_wmma_f32_16x16x32_f16(false, a1, false, b1, (short)0, c11, false, false);
      }
      // per-gate activation (waves 0-7:i, 8-15:f, 16-23:g(tanh), 24-31:o), store activated f16
      _Float16* gates = (_Float16*)sh_pool;
      const int gt = w >> 3;
      auto actstore = [&](v8f& c, int mt, int nt) {
        int n = nt * 16 + (lane & 15);
        #pragma unroll
        for (int r = 0; r < 8; ++r) {
          int b = mt * 16 + r + rb;
          float x = c[r];
          gates[b * NG + n] = (_Float16)((gt == 2) ? tanhf(x) : sigmoidf_(x));
        }
      };
      actstore(c00, 0, nt0); actstore(c01, 0, nt1); actstore(c10, 1, nt0); actstore(c11, 1, nt1);
    }
    __syncthreads();
    // ---- phase 3: LSTM combine ----
    {
      const _Float16* gates = (const _Float16*)sh_pool;
      for (int idx = tid; idx < NB * Hdim; idx += 1024) {
        int b = idx >> 8, j = idx & 255;
        float ai = (float)gates[b * NG + j];
        float af = (float)gates[b * NG + Hdim + j];
        float tg = (float)gates[b * NG + 2 * Hdim + j];
        float ao = (float)gates[b * NG + 3 * Hdim + j];
        float cc = af * sh_c[idx] + ai * tg;
        sh_c[idx] = cc;
        sh_h[idx] = (_Float16)(ao * tanhf(cc));
      }
    }
    __syncthreads();
    // ---- phase 4a: controller d,u (wave b, shfl_xor reduce) ----
    {
      int b = w;
      float sd = 0.f, su = 0.f;
      #pragma unroll
      for (int q = 0; q < 8; ++q) {
        float hv = (float)sh_h[b * Hdim + lane * 8 + q];
        sd += hv * Wd[lane * 8 + q];
        su += hv * Wu[lane * 8 + q];
      }
      for (int o = 16; o > 0; o >>= 1) { sd += __shfl_xor(sd, o); su += __shfl_xor(su, o); }
      if (lane == 0) { sh_du[b] = sigmoidf_(sd + bd[0]); sh_du[NB + b] = sigmoidf_(su + bu[0]); }
    }
    __syncthreads();
    // ---- phase 4b + 5: v = tanh(h@Wv^T+bv) and fused neural-stack scan (wave=batch, lane=m) ----
    {
      const int b = w, m = lane;
      float acc = bv[m];
      for (int k = 0; k < Hdim; ++k) acc += (float)sh_h[b * Hdim + k] * Wv[m * Hdim + k];
      const float vval = tanhf(acc);
      const float d = sh_du[b], u = sh_du[NB + b];
      Vbuf[((size_t)(t + 1) * NB + b) * Mdim + m] = vval;             // push at index t+1
      if (lane == 0) sbuf[(t + 1) * NB + b] = d;
      float tot_old = 0.f, tot_new = d;
      float r_acc = fminf(d, 1.0f) * vval;                            // coeff of pushed entry
      for (int l = t; l >= 0; --l) {
        float s_old = sbuf[l * NB + b];
        float sn = fmaxf(s_old - fmaxf(u - tot_old, 0.f), 0.f);
        tot_old += s_old;
        float coeff = fminf(sn, fmaxf(1.f - tot_new, 0.f));
        tot_new += sn;
        r_acc += coeff * Vbuf[((size_t)l * NB + b) * Mdim + m];
        if (lane == 0) sbuf[l * NB + b] = sn;
      }
      sh_r[b * Mdim + m] = (_Float16)r_acc;
    }
    __syncthreads();
    // ---- phase 6: decoder emission ----
    if (dec) {
      // pack h fragments (2 x 8)
      for (int idx = tid; idx < 16 * 512; idx += 1024) {
        int fr = idx >> 9, e = idx & 511, ln = e >> 4, j = e & 15;
        int mt = fr >> 3, kt = fr & 7;
        sh_frag[idx] = sh_h[(mt * 16 + (ln & 15)) * Hdim + kt * 32 + a_kmap(j, ln)];
      }
      __syncthreads();
      { // oc = tanh(h @ Wo^T + bo): 32 tiles, one per wave
        const v16h* wo = (const v16h*)(ws + OFF_WO);
        const v16h* af = (const v16h*)sh_frag;
        int mt = w >> 4, nt = w & 15;
        int n = nt * 16 + (lane & 15);
        v8f c; for (int r = 0; r < 8; ++r) c[r] = bo[n];
        for (int kt = 0; kt < 8; ++kt) {
          v16h a  = af[(mt * 8 + kt) * 32 + lane];
          v16h bb = wo[(kt * NTO + nt) * 32 + lane];
          c = __builtin_amdgcn_wmma_f32_16x16x32_f16(false, a, false, bb, (short)0, c, false, false);
        }
        _Float16* oc = (_Float16*)sh_pool;
        int rb = (lane >> 4) << 3;
        #pragma unroll
        for (int r = 0; r < 8; ++r) oc[(mt * 16 + r + rb) * Hdim + n] = (_Float16)tanhf(c[r]);
      }
      __syncthreads();
      { // pack oc fragments
        const _Float16* oc = (const _Float16*)sh_pool;
        for (int idx = tid; idx < 16 * 512; idx += 1024) {
          int fr = idx >> 9, e = idx & 511, ln = e >> 4, j = e & 15;
          int mt = fr >> 3, kt = fr & 7;
          sh_frag[idx] = oc[(mt * 16 + (ln & 15)) * Hdim + kt * 32 + a_kmap(j, ln)];
        }
      }
      __syncthreads();
      if (w < 18) { // logits = oc @ Wlin^T + blin: 2 x 9 tiles
        const v16h* wl = (const v16h*)(ws + OFF_WL);
        const v16h* af = (const v16h*)sh_frag;
        int mt = (w >= 9) ? 1 : 0, nt = w - mt * 9;
        int n = nt * 16 + (lane & 15);
        float bi = (n < OUTD) ? blin[n] : 0.0f;
        v8f c; for (int r = 0; r < 8; ++r) c[r] = bi;
        for (int kt = 0; kt < 8; ++kt) {
          v16h a  = af[(mt * 8 + kt) * 32 + lane];
          v16h bb = wl[(kt * NTL + nt) * 32 + lane];
          c = __builtin_amdgcn_wmma_f32_16x16x32_f16(false, a, false, bb, (short)0, c, false, false);
        }
        float* lg = (float*)(sh_pool + 32768);
        int rb = (lane >> 4) << 3;
        #pragma unroll
        for (int r = 0; r < 8; ++r) lg[(mt * 16 + r + rb) * NLPAD + n] = c[r];
      }
      __syncthreads();
      { // log_softmax row-wise (wave b), write output
        const float* lg = (const float*)(sh_pool + 32768);
        int b = w;
        float vals[5]; int cn = 0;
        float mv = -3.4e38f;
        for (int col = lane; col < OUTD; col += 32) { float x = lg[b * NLPAD + col]; vals[cn++] = x; mv = fmaxf(mv, x); }
        for (int o = 16; o > 0; o >>= 1) mv = fmaxf(mv, __shfl_xor(mv, o));
        float se = 0.f;
        for (int i = 0; i < cn; ++i) se += __expf(vals[i] - mv);
        for (int o = 16; o > 0; o >>= 1) se += __shfl_xor(se, o);
        float lz = __logf(se);
        float* orow = out + ((size_t)(t - TENC) * NB + b) * OUTD;
        int i = 0;
        for (int col = lane; col < OUTD; col += 32) orow[col] = vals[i++] - mv - lz;
      }
      __syncthreads();
    }
  }
}

// ======================= launcher =======================
extern "C" void kernel_launch(void* const* d_in, const int* in_sizes, int n_in,
                              void* d_out, int out_size, void* d_ws, size_t ws_size,
                              hipStream_t stream) {
  const float* X     = (const float*)d_in[0];
  const float* Y     = (const float*)d_in[1];
  const float* Wih_e = (const float*)d_in[2];
  const float* Whh_e = (const float*)d_in[3];
  const float* bih_e = (const float*)d_in[4];
  const float* bhh_e = (const float*)d_in[5];
  const float* Wih_d = (const float*)d_in[6];
  const float* Whh_d = (const float*)d_in[7];
  const float* bih_d = (const float*)d_in[8];
  const float* bhh_d = (const float*)d_in[9];
  const float* Wd    = (const float*)d_in[10];
  const float* bd    = (const float*)d_in[11];
  const float* Wu    = (const float*)d_in[12];
  const float* bu    = (const float*)d_in[13];
  const float* Wv    = (const float*)d_in[14];
  const float* bv    = (const float*)d_in[15];
  const float* Wo    = (const float*)d_in[16];
  const float* bo    = (const float*)d_in[17];
  const float* Wlin  = (const float*)d_in[18];
  const float* blin  = (const float*)d_in[19];
  const float* h0    = (const float*)d_in[20];
  const float* c0    = (const float*)d_in[21];
  unsigned char* ws  = (unsigned char*)d_ws;
  float* out         = (float*)d_out;

  stkrnn_pack_weights<<<512, 256, 0, stream>>>(Wih_e, Whh_e, Wih_d, Whh_d, Wo, Wlin, ws);
  stkrnn_xproj<<<4096, 256, 0, stream>>>(X, Y, bih_e, bhh_e, bih_d, bhh_d, ws);
  stkrnn_seq<<<1, 1024, 0, stream>>>(Wd, bd, Wu, bu, Wv, bv, bo, blin, h0, c0, ws, out);
}
